// MultiHeadAttention_54400055771505
// MI455X (gfx1250) — compile-verified, matrix operand order checked by reading, not run on hardware
//
#include <hip/hip_runtime.h>
#include <hip/hip_bf16.h>

typedef __bf16 bf16;
typedef bf16  v4bf  __attribute__((ext_vector_type(4)));
typedef bf16  v8bf  __attribute__((ext_vector_type(8)));
typedef bf16  v16bf __attribute__((ext_vector_type(16)));
typedef float v8f   __attribute__((ext_vector_type(8)));

#define B_   2
#define L_   2048
#define D_   1024
#define H_   16
#define HD_  64
#define M_   (B_ * L_)          // 4096 rows for the projection GEMMs
#define SROW 2052               // padded LDS row stride (floats) for scores/mean

// ---------------------------------------------------------------------------
// WMMA helpers (CDNA5 16x16x32 bf16 -> f32)
// ---------------------------------------------------------------------------
__device__ __forceinline__ v8f wmma_bf(v16bf a, v16bf b, v8f c) {
  return __builtin_amdgcn_wmma_f32_16x16x32_bf16(false, a, false, b, (short)0, c,
                                                 false, false);
}

__device__ __forceinline__ v16bf cat16(v8bf lo, v8bf hi) {
  return __builtin_shufflevector(lo, hi, 0, 1, 2, 3, 4, 5, 6, 7, 8, 9, 10, 11,
                                 12, 13, 14, 15);
}

// A/B fragment: lane owns row (A) / stored-row-as-column (B); a lane holds two
// contiguous 8-element K runs at k = 8*(lane/16) and +16 (ISA 7.12.2 layout).
__device__ __forceinline__ v16bf frag_bf16(const bf16* rp, int kb) {
  v8bf lo = *(const v8bf*)(rp + kb);
  v8bf hi = *(const v8bf*)(rp + kb + 16);
  return cat16(lo, hi);
}

__device__ __forceinline__ v16bf frag_f32(const float* rp, int kb) {
  v16bf r;
#pragma unroll
  for (int j = 0; j < 8; ++j) {
    r[j]     = (bf16)rp[kb + j];
    r[j + 8] = (bf16)rp[kb + 16 + j];
  }
  return r;
}

__device__ __forceinline__ v8f zero8() {
  v8f z = {0.f, 0.f, 0.f, 0.f, 0.f, 0.f, 0.f, 0.f};
  return z;
}

// ---------------------------------------------------------------------------
// CDNA5 async global->LDS copies (ASYNCcnt-tracked, no VGPR round trip)
// ---------------------------------------------------------------------------
__device__ __forceinline__ void async_lds_b128(unsigned ldsAddr, unsigned voff,
                                               const void* sbase) {
  asm volatile("global_load_async_to_lds_b128 %0, %1, %2"
               :
               : "v"(ldsAddr), "v"(voff), "s"((unsigned long long)sbase)
               : "memory");
}
__device__ __forceinline__ void async_lds_b64(unsigned ldsAddr, unsigned voff,
                                              const void* sbase) {
  asm volatile("global_load_async_to_lds_b64 %0, %1, %2"
               :
               : "v"(ldsAddr), "v"(voff), "s"((unsigned long long)sbase)
               : "memory");
}
__device__ __forceinline__ void wait_async0() {
  asm volatile("s_wait_asynccnt 0x0" ::: "memory");
}

// ---------------------------------------------------------------------------
// f32 -> bf16 conversion (8 elements / thread, packed 16B stores)
// ---------------------------------------------------------------------------
__global__ __launch_bounds__(256) void cvt_bf16(const float* __restrict__ s,
                                                bf16* __restrict__ d, int n) {
  const int i = (blockIdx.x * 256 + threadIdx.x) * 8;
  if (i + 8 <= n) {
    v8bf o;
#pragma unroll
    for (int j = 0; j < 8; ++j) o[j] = (bf16)s[i + j];
    *(v8bf*)(d + i) = o;
  }
}

// ---------------------------------------------------------------------------
// GEMM: out = A @ W^T + bias (A, W already bf16).
// mode 0/1: bf16 [B,H,L,HD] (Q/K)   mode 2: bf16 [B,H,HD,L] (V^T)
// mode 3:   f32  [M,D] -> d_out     (output projection)
// 256 thr (8 waves = 4Mx2N), 128x128 tile, K chunk 32, double-buffered LDS
// filled by GLOBAL_LOAD_ASYNC_TO_LDS_B128.
// ---------------------------------------------------------------------------
__global__ __launch_bounds__(256) void gemm_bf16(const bf16* __restrict__ A,
                                                 const bf16* __restrict__ Wb,
                                                 const float* __restrict__ bias,
                                                 void* __restrict__ outAny,
                                                 int mode) {
  __shared__ bf16 sA[2][128 * 40];
  __shared__ bf16 sB[2][128 * 40];

  const int t    = threadIdx.x;
  const int wave = t >> 5;
  const int lane = t & 31;
  const int wm   = wave & 3;   // 4 wave rows (32 M each)
  const int wn   = wave >> 2;  // 2 wave cols (64 N each)
  const int lr   = lane & 15;
  const int lh   = lane >> 4;
  const int kb   = lh * 8;

  const int m0 = blockIdx.x * 128;
  const int n0 = blockIdx.y * 128;

  v8f acc[2][4];
#pragma unroll
  for (int mt = 0; mt < 2; ++mt)
#pragma unroll
    for (int nt = 0; nt < 4; ++nt) acc[mt][nt] = zero8();

  // staging: thread owns rows r0 and r0+64, 16B column segment s0
  const int r0 = t >> 2;
  const int s0 = (t & 3) * 8;
  unsigned aoff0 = (unsigned)(((m0 + r0) * D_ + s0) * 2);
  unsigned aoff1 = aoff0 + 64u * D_ * 2u;
  unsigned boff0 = (unsigned)(((n0 + r0) * D_ + s0) * 2);
  unsigned boff1 = boff0 + 64u * D_ * 2u;
  unsigned lA0[2], lA1[2], lB0[2], lB1[2];
#pragma unroll
  for (int p = 0; p < 2; ++p) {
    lA0[p] = (unsigned)(size_t)&sA[p][r0 * 40 + s0];
    lA1[p] = (unsigned)(size_t)&sA[p][(r0 + 64) * 40 + s0];
    lB0[p] = (unsigned)(size_t)&sB[p][r0 * 40 + s0];
    lB1[p] = (unsigned)(size_t)&sB[p][(r0 + 64) * 40 + s0];
  }

  // prologue: chunk 0 -> buffer 0
  async_lds_b128(lA0[0], aoff0, A);
  async_lds_b128(lA1[0], aoff1, A);
  async_lds_b128(lB0[0], boff0, Wb);
  async_lds_b128(lB1[0], boff1, Wb);
  aoff0 += 64; aoff1 += 64; boff0 += 64; boff1 += 64;
  wait_async0();
  __syncthreads();

  for (int kc = 0; kc < D_; kc += 32) {
    const int p = (kc >> 5) & 1;
    if (kc + 32 < D_) {  // prefetch next chunk into the other buffer
      async_lds_b128(lA0[p ^ 1], aoff0, A);
      async_lds_b128(lA1[p ^ 1], aoff1, A);
      async_lds_b128(lB0[p ^ 1], boff0, Wb);
      async_lds_b128(lB1[p ^ 1], boff1, Wb);
      aoff0 += 64; aoff1 += 64; boff0 += 64; boff1 += 64;
    }

    v16bf af[2], bfg[4];
#pragma unroll
    for (int mt = 0; mt < 2; ++mt)
      af[mt] = frag_bf16(&sA[p][(wm * 32 + mt * 16 + lr) * 40], kb);
#pragma unroll
    for (int nt = 0; nt < 4; ++nt)
      bfg[nt] = frag_bf16(&sB[p][(wn * 64 + nt * 16 + lr) * 40], kb);
#pragma unroll
    for (int mt = 0; mt < 2; ++mt)
#pragma unroll
      for (int nt = 0; nt < 4; ++nt)
        acc[mt][nt] = wmma_bf(af[mt], bfg[nt], acc[mt][nt]);

    wait_async0();
    __syncthreads();
  }

  // ---- epilogue: bias add + layout-specific scatter ----
#pragma unroll
  for (int mt = 0; mt < 2; ++mt)
#pragma unroll
    for (int nt = 0; nt < 4; ++nt) {
      const int   n     = n0 + wn * 64 + nt * 16 + lr;
      const float bb    = bias[n];
      const int   mbase = m0 + wm * 32 + mt * 16 + lh * 8;
#pragma unroll
      for (int v = 0; v < 8; ++v) {
        const int   m   = mbase + v;
        const float val = acc[mt][nt][v] + bb;
        if (mode == 3) {
          ((float*)outAny)[(size_t)m * D_ + n] = val;
        } else {
          const int bi = m >> 11, l = m & (L_ - 1);
          const int h = n >> 6, hd = n & (HD_ - 1);
          size_t idx;
          if (mode == 2)
            idx = ((size_t)((bi * H_ + h) * HD_ + hd)) * L_ + l;  // V transposed
          else
            idx = ((size_t)((bi * H_ + h) * L_ + l)) * HD_ + hd;  // Q / K
          ((bf16*)outAny)[idx] = (bf16)val;
        }
      }
    }
}

// ---------------------------------------------------------------------------
// Attention: one block per (b, 16-row q tile); serial head loop (deterministic
// mean accumulation). scores[16][SROW] + mean[16][SROW] fp32 in dynamic LDS.
// ---------------------------------------------------------------------------
#define SMEM_BYTES (16 * SROW * 4 * 2 + 16 * HD_ * 2 + 4 * 256 * 4)  // 268800

__global__ __launch_bounds__(256) void attn_kernel(
    const bf16* __restrict__ Qb, const bf16* __restrict__ Kb,
    const bf16* __restrict__ Vt, bf16* __restrict__ Att,
    float* __restrict__ meanOut) {
  extern __shared__ char smemRaw[];
  float* scores = (float*)smemRaw;             // 16 x SROW
  float* meanS  = scores + 16 * SROW;          // 16 x SROW
  bf16*  qs     = (bf16*)(meanS + 16 * SROW);  // 16 x 64
  float* red    = (float*)(qs + 16 * HD_);     // 4 x 256 partial reduce

  const int t    = threadIdx.x;
  const int wave = t >> 5;
  const int lane = t & 31;
  const int lr   = lane & 15;
  const int lh   = lane >> 4;
  const int kb   = lh * 8;

  const int qt = blockIdx.x;  // 0..127 (16-row q tile)
  const int b  = blockIdx.y;  // 0..1
  const int q0 = qt * 16;

  // zero the mean accumulator (consumed after first barrier)
  for (int i = t; i < 16 * SROW; i += 256) meanS[i] = 0.f;

  const int row     = t >> 4;  // softmax: 16 threads per row
  const int colLane = t & 15;

  const unsigned qsLds = (unsigned)(size_t)(qs + t * 4);

  for (int h = 0; h < H_; ++h) {
    // ---- async-copy Q tile [16,64] bf16 straight into LDS ----
    {
      const unsigned off =
          (unsigned)(((((b * H_ + h) * L_) + q0) * HD_ + t * 4) * 2);
      async_lds_b64(qsLds, off, Qb);
      wait_async0();
    }
    __syncthreads();

    // ---- scores = (Q K^T) * 1/sqrt(HD): wave w owns columns [w*256,+256) ----
    {
      v16bf aq0 = frag_bf16(&qs[lr * HD_], kb);       // K dims 0..31
      v16bf aq1 = frag_bf16(&qs[lr * HD_ + 32], kb);  // K dims 32..63
#pragma unroll 4
      for (int nt = 0; nt < 16; ++nt) {
        const int   c0 = (wave * 16 + nt) * 16;
        const bf16* kp = Kb + (((size_t)(b * H_ + h)) * L_ + c0 + lr) * HD_;
        v16bf bk0 = frag_bf16(kp, kb);
        v16bf bk1 = frag_bf16(kp + 32, kb);
        v8f s = zero8();
        s = wmma_bf(aq0, bk0, s);
        s = wmma_bf(aq1, bk1, s);
        const int c = c0 + lr;
#pragma unroll
        for (int v = 0; v < 8; ++v)
          scores[(v + 8 * lh) * SROW + c] = s[v] * 0.125f;  // HD^-0.5
      }
    }
    __syncthreads();

    // ---- softmax over Lk (16 threads per row) + mean accumulation ----
    {
      float* srow = scores + row * SROW;
      float* mrow = meanS + row * SROW;
      float  mx   = -3.4e38f;
      for (int j = 0; j < 128; ++j) mx = fmaxf(mx, srow[colLane + 16 * j]);
#pragma unroll
      for (int d = 1; d < 16; d <<= 1) mx = fmaxf(mx, __shfl_xor(mx, d, 32));
      float sum = 0.f;
      for (int j = 0; j < 128; ++j) {
        const int c = colLane + 16 * j;
        float e = __expf(srow[c] - mx);
        srow[c] = e;
        sum += e;
      }
#pragma unroll
      for (int d = 1; d < 16; d <<= 1) sum += __shfl_xor(sum, d, 32);
      const float inv = 1.f / sum;
      for (int j = 0; j < 128; ++j) {
        const int c = colLane + 16 * j;
        float wv = srow[c] * inv;
        srow[c] = wv;
        mrow[c] += wv * 0.0625f;  // 1/H
      }
    }
    __syncthreads();

    // ---- attended = weights @ V : 4 N tiles x 2 K halves over 8 waves ----
    {
      const int wn = wave & 3;   // hd tile
      const int kh = wave >> 2;  // K half (1024 each)
      v8f a = zero8();
      const bf16* vb =
          Vt + (((size_t)(b * H_ + h)) * HD_ + wn * 16 + lr) * L_ + kh * 1024;
      const float* sr2 = scores + lr * SROW + kh * 1024;
      for (int kc = 0; kc < 1024; kc += 32) {
        v16bf aw  = frag_f32(sr2 + kc, kb);
        v16bf bv2 = frag_bf16(vb + kc, kb);
        a = wmma_bf(aw, bv2, a);
      }
      if (kh == 1) {
#pragma unroll
        for (int v = 0; v < 8; ++v)
          red[wn * 256 + (v + 8 * lh) * 16 + lr] = a[v];
      }
      __syncthreads();
      if (kh == 0) {
#pragma unroll
        for (int v = 0; v < 8; ++v) {
          const float val = a[v] + red[wn * 256 + (v + 8 * lh) * 16 + lr];
          const int   lq  = q0 + v + 8 * lh;
          Att[((size_t)(b * L_ + lq)) * D_ + h * HD_ + wn * 16 + lr] = (bf16)val;
        }
      }
      __syncthreads();  // scores/qs reused next head
    }
  }

  // ---- write weights.mean(axis=heads) ----
  float* mo = meanOut + ((size_t)(b * L_ + q0)) * L_;
  for (int i = t; i < 16 * L_; i += 256) {
    const int rr = i >> 11, cc = i & (L_ - 1);
    mo[(size_t)rr * L_ + cc] = meanS[rr * SROW + cc];
  }
}

// ---------------------------------------------------------------------------
// Launch
// ---------------------------------------------------------------------------
extern "C" void kernel_launch(void* const* d_in, const int* in_sizes, int n_in,
                              void* d_out, int out_size, void* d_ws,
                              size_t ws_size, hipStream_t stream) {
  const float* q  = (const float*)d_in[0];
  const float* k  = (const float*)d_in[1];
  const float* v  = (const float*)d_in[2];
  const float* Wq = (const float*)d_in[3];
  const float* bq = (const float*)d_in[4];
  const float* Wk = (const float*)d_in[5];
  const float* bk = (const float*)d_in[6];
  const float* Wv = (const float*)d_in[7];
  const float* bv = (const float*)d_in[8];
  const float* Wo = (const float*)d_in[9];
  const float* bo = (const float*)d_in[10];

  const size_t MB = (size_t)1 << 20;
  char* ws = (char*)d_ws;
  bf16* Xq  = (bf16*)(ws);             // 8 MiB  bf16 copy of query
  bf16* Xk  = (bf16*)(ws + 8 * MB);    // 8 MiB  bf16 copy of key
  bf16* Xv  = (bf16*)(ws + 16 * MB);   // 8 MiB  bf16 copy of value
  bf16* Wqb = (bf16*)(ws + 24 * MB);   // 2 MiB
  bf16* Wkb = (bf16*)(ws + 26 * MB);   // 2 MiB
  bf16* Wvb = (bf16*)(ws + 28 * MB);   // 2 MiB
  bf16* Wob = (bf16*)(ws + 30 * MB);   // 2 MiB
  bf16* Qb  = (bf16*)(ws + 32 * MB);   // 8 MiB  [B,H,L,HD]
  bf16* Kb  = (bf16*)(ws + 40 * MB);   // 8 MiB  [B,H,L,HD]
  bf16* Vt  = (bf16*)(ws + 48 * MB);   // 8 MiB  [B,H,HD,L]
  bf16* Att = (bf16*)(ws + 56 * MB);   // 8 MiB  [B,L,D]

  float* out     = (float*)d_out;
  float* meanOut = out + (size_t)M_ * D_;

  (void)hipFuncSetAttribute((const void*)attn_kernel,
                            hipFuncAttributeMaxDynamicSharedMemorySize,
                            SMEM_BYTES);

  const int nX = M_ * D_;  // 4,194,304
  const int nW = D_ * D_;  // 1,048,576
  cvt_bf16<<<nX / (8 * 256), 256, 0, stream>>>(q, Xq, nX);
  cvt_bf16<<<nX / (8 * 256), 256, 0, stream>>>(k, Xk, nX);
  cvt_bf16<<<nX / (8 * 256), 256, 0, stream>>>(v, Xv, nX);
  cvt_bf16<<<nW / (8 * 256), 256, 0, stream>>>(Wq, Wqb, nW);
  cvt_bf16<<<nW / (8 * 256), 256, 0, stream>>>(Wk, Wkb, nW);
  cvt_bf16<<<nW / (8 * 256), 256, 0, stream>>>(Wv, Wvb, nW);
  cvt_bf16<<<nW / (8 * 256), 256, 0, stream>>>(Wo, Wob, nW);

  dim3 gg(M_ / 128, D_ / 128);
  gemm_bf16<<<gg, 256, 0, stream>>>(Xq, Wqb, bq, Qb, 0);
  gemm_bf16<<<gg, 256, 0, stream>>>(Xk, Wkb, bk, Kb, 1);
  gemm_bf16<<<gg, 256, 0, stream>>>(Xv, Wvb, bv, Vt, 2);
  attn_kernel<<<dim3(L_ / 16, B_), 256, SMEM_BYTES, stream>>>(Qb, Kb, Vt, Att,
                                                              meanOut);
  gemm_bf16<<<gg, 256, 0, stream>>>(Att, Wob, bo, out, 3);
}
